// MultiHeadClusterAttention_7078106104047
// MI455X (gfx1250) — matrix operand order repaired and checked by reading, not compiled, not run on hardware
//
#include <hip/hip_runtime.h>

// ---------------------------------------------------------------------------
// MultiHeadClusterAttention on MI455X (gfx1250, wave32, WMMA)
// ---------------------------------------------------------------------------
typedef unsigned short u16;
typedef __attribute__((ext_vector_type(16))) __bf16 bf16x16;
typedef __attribute__((ext_vector_type(8)))  float  f32x8;

constexpr int N_   = 32768;   // tokens
constexpr int E_   = 512;     // embedding
constexpr int H_   = 8;       // heads
constexpr int D_   = 64;      // head dim
constexpr int KC_  = 327;     // clusters (n / 100)
constexpr int KCP_ = 352;     // clusters padded to 32
constexpr int NT_  = KCP_/16; // 22 cluster tiles
constexpr int IT_  = 10;      // kmeans iterations

union FragU { bf16x16 v; uint4 q[2]; };

__device__ __forceinline__ u16 f2bf(float f) {
  union { float f; unsigned u; } x; x.f = f;
  unsigned r = x.u + 0x7FFFu + ((x.u >> 16) & 1u);   // round-to-nearest-even
  return (u16)(r >> 16);
}

// A-fragment 16x32 (MxK): lanes 0-15 row m, K 0-7 & 16-23; lanes 16-31 K 8-15 & 24-31.
__device__ __forceinline__ bf16x16 load_frag_a(const u16* p, int m0, int ld, int k0, int lane) {
  int m  = m0 + (lane & 15);
  int kb = k0 + ((lane >> 4) << 3);
  const u16* r = p + (size_t)m * ld + kb;
  FragU f;
  f.q[0] = *(const uint4*)(r);
  f.q[1] = *(const uint4*)(r + 16);
  return f.v;
}

// B-fragment 32x16 (KxN) from Bt rows (element (k,n) = Bt[n*ld + k]):
// lanes 0-15 col n, K 0-7 & 16-23; lanes 16-31 K 8-15 & 24-31.
__device__ __forceinline__ bf16x16 load_frag_b(const u16* p, int n0, int ld, int k0, int lane) {
  int n  = n0 + (lane & 15);
  int kb = k0 + ((lane >> 4) << 3);
  const u16* r = p + (size_t)n * ld + kb;
  FragU f;
  f.q[0] = *(const uint4*)(r);
  f.q[1] = *(const uint4*)(r + 16);
  return f.v;
}

__device__ __forceinline__ f32x8 wmma_bf16(bf16x16 a, bf16x16 b, f32x8 c) {
  return __builtin_amdgcn_wmma_f32_16x16x32_bf16(false, a, false, b, (short)0, c, false, false);
}

// ---------------------------------------------------------------------------
// Elementwise helpers
// ---------------------------------------------------------------------------
__global__ void cvt_f32_bf16(const float* __restrict__ src, u16* __restrict__ dst, int n) {
  int i = blockIdx.x * blockDim.x + threadIdx.x;
  if (i < n) dst[i] = f2bf(src[i]);
}

__global__ void zero_f32(float* __restrict__ ptr, int n) {
  int i = blockIdx.x * blockDim.x + threadIdx.x;
  if (i < n) ptr[i] = 0.0f;
}

// ---------------------------------------------------------------------------
// WMMA GEMM (K = lda = ldb = ldc = 512, compile time):
//   C[m][n] = sum_k A[m][k] * Bt[n][k] + bias[n]
// block = 128 thr (4 waves); wave -> 16x64 tile; block -> 64x64.
// Fully unrolled + software pipelined: straight-line SSA lets the register
// allocator rename the ping-pong buffers (no v_dual_mov rotation) while the
// k+32 loads stay hoisted above k's WMMAs (partial s_wait_loadcnt overlap).
// ---------------------------------------------------------------------------
__global__ __launch_bounds__(128) void gemm_tn_bf16(
    const u16* __restrict__ A, const u16* __restrict__ Bt,
    const float* __restrict__ bias,
    u16* __restrict__ Cb, float* __restrict__ Cf, int M)
{
  constexpr int KSTEPS = E_ / 32;   // 16
  int lane = threadIdx.x & 31;
  int wave = threadIdx.x >> 5;
  int m0 = blockIdx.y * 64 + wave * 16;
  int n0 = blockIdx.x * 64;
  if (m0 >= M) return;

  f32x8 acc[4] = {};
  bf16x16 a_cur = load_frag_a(A, m0, E_, 0, lane);
  bf16x16 b_cur[4];
#pragma unroll
  for (int j = 0; j < 4; ++j) b_cur[j] = load_frag_b(Bt, n0 + j * 16, E_, 0, lane);

#pragma unroll
  for (int ks = 0; ks < KSTEPS; ++ks) {
    const int k0 = ks * 32;
    bf16x16 a_nxt = a_cur;
    bf16x16 b_nxt[4] = {b_cur[0], b_cur[1], b_cur[2], b_cur[3]};
    if (ks + 1 < KSTEPS) {
      a_nxt = load_frag_a(A, m0, E_, k0 + 32, lane);
#pragma unroll
      for (int j = 0; j < 4; ++j) b_nxt[j] = load_frag_b(Bt, n0 + j * 16, E_, k0 + 32, lane);
    }
#pragma unroll
    for (int j = 0; j < 4; ++j) acc[j] = wmma_bf16(a_cur, b_cur[j], acc[j]);
    a_cur = a_nxt;
#pragma unroll
    for (int j = 0; j < 4; ++j) b_cur[j] = b_nxt[j];
  }

  int row_hi = (lane >> 4) << 3;
  int col = lane & 15;
#pragma unroll
  for (int j = 0; j < 4; ++j) {
    float bv = bias ? bias[n0 + j * 16 + col] : 0.0f;
#pragma unroll
    for (int r = 0; r < 8; ++r) {
      float v = acc[j][r] + bv;
      size_t idx = (size_t)(m0 + row_hi + r) * E_ + (n0 + j * 16 + col);
      if (Cf) Cf[idx] = v;
      if (Cb) Cb[idx] = f2bf(v);
    }
  }
}

// ---------------------------------------------------------------------------
// KMeans
// ---------------------------------------------------------------------------
__global__ __launch_bounds__(256) void kmeans_init(const float* __restrict__ x,
                                                   float* __restrict__ cents) {
  int c = blockIdx.x;  // [0, KCP_)
  for (int j = threadIdx.x; j < E_; j += blockDim.x) {
    float v = 0.0f;
    if (c < KC_) {
      long long idx = ((long long)c * (N_ - 1)) / (KC_ - 1);  // linspace init
      v = x[idx * E_ + j];
    }
    cents[(size_t)c * E_ + j] = v;
  }
}

__global__ __launch_bounds__(256) void cent_prep(const float* __restrict__ cents,
                                                 u16* __restrict__ cb,
                                                 float* __restrict__ c2) {
  __shared__ float red[256];
  int c = blockIdx.x;
  float s = 0.0f;
  for (int j = threadIdx.x; j < E_; j += 256) {
    float v = cents[(size_t)c * E_ + j];
    cb[(size_t)c * E_ + j] = f2bf(v);
    s += v * v;
  }
  red[threadIdx.x] = s;
  __syncthreads();
  for (int off = 128; off; off >>= 1) {
    if (threadIdx.x < (unsigned)off) red[threadIdx.x] += red[threadIdx.x + off];
    __syncthreads();
  }
  if (threadIdx.x == 0) c2[c] = (c < KC_) ? red[0] : 1e30f;
}

// Assignment: wave computes S^T[c][i] = c . x_i via WMMA (clusters = M dim),
// argmin_c (|c|^2 - 2 c.x) per point.  block = 8 waves = 128 points.
// The 16 x-fragments are invariant across all 22 cluster tiles -> preload
// them once into 128 VGPRs (wave32 budget) instead of reloading 22x.
__global__ __launch_bounds__(256) void kmeans_assign(
    const u16* __restrict__ xb, const u16* __restrict__ cb,
    const float* __restrict__ c2, int* __restrict__ asg)
{
  int lane = threadIdx.x & 31;
  int wave = threadIdx.x >> 5;
  int pt0 = (blockIdx.x * 8 + wave) * 16;
  int row_hi = (lane >> 4) << 3;

  bf16x16 xf[E_ / 32];
#pragma unroll
  for (int kk = 0; kk < E_ / 32; ++kk)
    xf[kk] = load_frag_b(xb, pt0, E_, kk * 32, lane);

  float best = 3e38f;
  int bidx = 0;
  for (int t = 0; t < NT_; ++t) {
    f32x8 acc = {};
#pragma unroll
    for (int kk = 0; kk < E_ / 32; ++kk) {
      bf16x16 a = load_frag_a(cb, t * 16, E_, kk * 32, lane);
      acc = wmma_bf16(a, xf[kk], acc);
    }
#pragma unroll
    for (int r = 0; r < 8; ++r) {
      int g = t * 16 + row_hi + r;
      float d = c2[g] - 2.0f * acc[r];     // padded rows have c2 = 1e30
      if (d < best || (d == best && g < bidx)) { best = d; bidx = g; }
    }
  }
  float ob = __shfl_xor(best, 16, 32);
  int   oi = __shfl_xor(bidx, 16, 32);
  if (ob < best || (ob == best && oi < bidx)) { best = ob; bidx = oi; }
  if (lane < 16) asg[pt0 + lane] = bidx;
}

__global__ __launch_bounds__(256) void kmeans_update(
    const float* __restrict__ x, const int* __restrict__ asg,
    float* __restrict__ sums, float* __restrict__ cnt)
{
  int i = blockIdx.x * blockDim.x + threadIdx.x;
  if (i >= N_) return;
  int c = asg[i];
  const float4* xr = (const float4*)(x + (size_t)i * E_);
  float* s = sums + (size_t)c * E_;
  for (int j = 0; j < E_ / 4; ++j) {
    float4 v = xr[j];
    atomicAdd(s + j * 4 + 0, v.x);
    atomicAdd(s + j * 4 + 1, v.y);
    atomicAdd(s + j * 4 + 2, v.z);
    atomicAdd(s + j * 4 + 3, v.w);
  }
  atomicAdd(cnt + c, 1.0f);
}

__global__ void cent_new(const float* __restrict__ sums, const float* __restrict__ cnt,
                         float* __restrict__ cents) {
  int idx = blockIdx.x * blockDim.x + threadIdx.x;
  if (idx >= KC_ * E_) return;        // padded rows stay zero
  int c = idx / E_;
  float ct = cnt[c];
  if (ct > 0.5f) cents[idx] = sums[idx] / ct;   // else keep old centroid
}

__global__ void transpose_v(const u16* __restrict__ vb, u16* __restrict__ vt) {
  int idx = blockIdx.x * blockDim.x + threadIdx.x;   // h*D*KCP + d*KCP + c
  if (idx >= H_ * D_ * KCP_) return;
  int c = idx % KCP_;
  int hd = idx / KCP_;
  vt[idx] = vb[(size_t)c * E_ + hd];
}

// ---------------------------------------------------------------------------
// Fused attention: wave = 16 queries x 1 head; all 352 (padded) keys.
//   1) S^T[c][i] via WMMA (keys = M dim, queries = N dim), 44 WMMAs
//      (q-fragments hoisted: invariant across the 22 key tiles)
//   2) masked softmax over clusters (VGPR rows + shfl_xor(16) across halves)
//   3) att (bf16, /sqrt(512) folded in) staged to wave-private LDS
//   4) out[i][d] = att . V via WMMA (44 more, software-pipelined), bf16 out
// ---------------------------------------------------------------------------
__global__ __launch_bounds__(128) void attention_fused(
    const u16* __restrict__ qb, const u16* __restrict__ kb,
    const u16* __restrict__ vt, u16* __restrict__ outb)
{
  __shared__ __align__(16) u16 attb[4][16 * KCP_];
  int lane = threadIdx.x & 31;
  int wave = threadIdx.x >> 5;
  int unit = blockIdx.x * 4 + wave;        // 16384 units
  int h   = unit & (H_ - 1);
  int pt0 = (unit >> 3) * 16;
  int row_hi = (lane >> 4) << 3;
  int col = lane & 15;

  // ---- scores ----
  bf16x16 qf[2];
#pragma unroll
  for (int kk = 0; kk < 2; ++kk)
    qf[kk] = load_frag_b(qb, pt0, E_, h * D_ + kk * 32, lane);

  f32x8 sacc[NT_] = {};
#pragma unroll
  for (int t = 0; t < NT_; ++t) {
#pragma unroll
    for (int kk = 0; kk < 2; ++kk) {
      bf16x16 a = load_frag_a(kb, t * 16, E_, h * D_ + kk * 32, lane);
      sacc[t] = wmma_bf16(a, qf[kk], sacc[t]);
    }
  }

  // ---- masked softmax over clusters ----
  float mx = -3e38f;
#pragma unroll
  for (int t = 0; t < NT_; ++t)
#pragma unroll
    for (int r = 0; r < 8; ++r) {
      int g = t * 16 + row_hi + r;
      float s = (g < KC_) ? sacc[t][r] : -3e38f;
      sacc[t][r] = s;
      mx = fmaxf(mx, s);
    }
  mx = fmaxf(mx, __shfl_xor(mx, 16, 32));
  float sum = 0.0f;
#pragma unroll
  for (int t = 0; t < NT_; ++t)
#pragma unroll
    for (int r = 0; r < 8; ++r) {
      float e = __expf(sacc[t][r] - mx);
      sacc[t][r] = e;
      sum += e;
    }
  sum += __shfl_xor(sum, 16, 32);
  float scale = 0.044194173824159216f / sum;   // (1/sqrt(512)) applied AFTER softmax

  u16* arow = &attb[wave][0];                  // wave-private: no barrier needed
#pragma unroll
  for (int t = 0; t < NT_; ++t)
#pragma unroll
    for (int r = 0; r < 8; ++r) {
      int g = t * 16 + row_hi + r;
      arow[col * KCP_ + g] = f2bf(sacc[t][r] * scale);   // pads -> 0
    }

  // ---- out = att @ V  (double-buffered over k; fully unrolled) ----
  const u16* vth = vt + (size_t)h * D_ * KCP_;
  f32x8 oacc[4] = {};
  bf16x16 a_cur = load_frag_a(arow, 0, KCP_, 0, lane);   // LDS reads
  bf16x16 b_cur[4];
#pragma unroll
  for (int j = 0; j < 4; ++j) b_cur[j] = load_frag_b(vth, j * 16, KCP_, 0, lane);
#pragma unroll
  for (int ks = 0; ks < NT_ / 2; ++ks) {
    const int k0 = ks * 32;
    bf16x16 a_nxt = a_cur;
    bf16x16 b_nxt[4] = {b_cur[0], b_cur[1], b_cur[2], b_cur[3]};
    if (ks + 1 < NT_ / 2) {
      a_nxt = load_frag_a(arow, 0, KCP_, k0 + 32, lane);
#pragma unroll
      for (int j = 0; j < 4; ++j) b_nxt[j] = load_frag_b(vth, j * 16, KCP_, k0 + 32, lane);
    }
#pragma unroll
    for (int j = 0; j < 4; ++j) oacc[j] = wmma_bf16(a_cur, b_cur[j], oacc[j]);
    a_cur = a_nxt;
#pragma unroll
    for (int j = 0; j < 4; ++j) b_cur[j] = b_nxt[j];
  }
#pragma unroll
  for (int j = 0; j < 4; ++j)
#pragma unroll
    for (int r = 0; r < 8; ++r)
      outb[(size_t)(pt0 + row_hi + r) * E_ + h * D_ + j * 16 + col] = f2bf(oacc[j][r]);
}

// ---------------------------------------------------------------------------
// Host-side orchestration
// ---------------------------------------------------------------------------
extern "C" void kernel_launch(void* const* d_in, const int* in_sizes, int n_in,
                              void* d_out, int out_size, void* d_ws, size_t ws_size,
                              hipStream_t stream) {
  const float* x  = (const float*)d_in[0];
  const float* Wq = (const float*)d_in[2];
  const float* bq = (const float*)d_in[3];
  const float* Wk = (const float*)d_in[4];
  const float* bk = (const float*)d_in[5];
  const float* Wv = (const float*)d_in[6];
  const float* bv = (const float*)d_in[7];
  const float* Wo = (const float*)d_in[8];
  const float* bo = (const float*)d_in[9];
  float* out = (float*)d_out;
  (void)in_sizes; (void)n_in; (void)out_size; (void)ws_size;

  char* base = (char*)d_ws;
  size_t off = 0;
  auto take = [&](size_t bytes) -> void* {
    void* r = base + off;
    off += (bytes + 255) & ~(size_t)255;
    return r;
  };
  u16*   xb    = (u16*)  take((size_t)N_ * E_ * 2);
  u16*   qb    = (u16*)  take((size_t)N_ * E_ * 2);
  u16*   outb  = (u16*)  take((size_t)N_ * E_ * 2);
  u16*   wqb   = (u16*)  take((size_t)E_ * E_ * 2);
  u16*   wkb   = (u16*)  take((size_t)E_ * E_ * 2);
  u16*   wvb   = (u16*)  take((size_t)E_ * E_ * 2);
  u16*   wob   = (u16*)  take((size_t)E_ * E_ * 2);
  float* cents = (float*)take((size_t)KCP_ * E_ * 4);
  u16*   cb    = (u16*)  take((size_t)KCP_ * E_ * 2);
  float* c2    = (float*)take((size_t)KCP_ * 4);
  float* sums  = (float*)take((size_t)KCP_ * E_ * 4);
  float* cnt   = (float*)take((size_t)KCP_ * 4);
  int*   asg   = (int*)  take((size_t)N_ * 4);
  u16*   kbuf  = (u16*)  take((size_t)KCP_ * E_ * 2);
  u16*   vbuf  = (u16*)  take((size_t)KCP_ * E_ * 2);
  u16*   vt    = (u16*)  take((size_t)H_ * D_ * KCP_ * 2);

  // 1) bf16 conversions
  cvt_f32_bf16<<<(N_ * E_ + 255) / 256, 256, 0, stream>>>(x, xb, N_ * E_);
  cvt_f32_bf16<<<(E_ * E_ + 255) / 256, 256, 0, stream>>>(Wq, wqb, E_ * E_);
  cvt_f32_bf16<<<(E_ * E_ + 255) / 256, 256, 0, stream>>>(Wk, wkb, E_ * E_);
  cvt_f32_bf16<<<(E_ * E_ + 255) / 256, 256, 0, stream>>>(Wv, wvb, E_ * E_);
  cvt_f32_bf16<<<(E_ * E_ + 255) / 256, 256, 0, stream>>>(Wo, wob, E_ * E_);

  // 2) Q projection
  gemm_tn_bf16<<<dim3(E_ / 64, N_ / 64), 128, 0, stream>>>(xb, wqb, bq, qb, nullptr, N_);

  // 3) KMeans
  kmeans_init<<<KCP_, 256, 0, stream>>>(x, cents);
  cent_prep<<<KCP_, 256, 0, stream>>>(cents, cb, c2);
  for (int it = 0; it < IT_; ++it) {
    zero_f32<<<(KCP_ * E_ + 255) / 256, 256, 0, stream>>>(sums, KCP_ * E_);
    zero_f32<<<2, 256, 0, stream>>>(cnt, KCP_);
    kmeans_assign<<<N_ / 128, 256, 0, stream>>>(xb, cb, c2, asg);
    kmeans_update<<<N_ / 256, 256, 0, stream>>>(x, asg, sums, cnt);
    cent_new<<<(KC_ * E_ + 255) / 256, 256, 0, stream>>>(sums, cnt, cents);
    cent_prep<<<KCP_, 256, 0, stream>>>(cents, cb, c2);
  }

  // 4) K / V projections on centroids (+ V transpose per head)
  gemm_tn_bf16<<<dim3(E_ / 64, (KCP_ + 63) / 64), 128, 0, stream>>>(cb, wkb, bk, kbuf, nullptr, KCP_);
  gemm_tn_bf16<<<dim3(E_ / 64, (KCP_ + 63) / 64), 128, 0, stream>>>(cb, wvb, bv, vbuf, nullptr, KCP_);
  transpose_v<<<(H_ * D_ * KCP_ + 255) / 256, 256, 0, stream>>>(vbuf, vt);

  // 5) fused attention (softmax / sqrt(E) folded into attention weights)
  attention_fused<<<(N_ / 16) * H_ / 4, 128, 0, stream>>>(qb, kbuf, vt, outb);

  // 6) output projection -> f32 result
  gemm_tn_bf16<<<dim3(E_ / 64, N_ / 64), 128, 0, stream>>>(outb, wob, bo, nullptr, out, N_);
}
